// Label_wise_attention_75943611728722
// MI455X (gfx1250) — matrix operand order repaired and checked by reading, not compiled, not run on hardware
//
#include <hip/hip_runtime.h>
#include <hip/hip_bf16.h>
#include <math.h>

#define B_ 8
#define S_ 512
#define D_ 768
#define E_ 768
#define L_ 4000

typedef __attribute__((ext_vector_type(16))) _Float16 v16h;
typedef __attribute__((ext_vector_type(8)))  _Float16 v8h;
typedef __attribute__((ext_vector_type(8)))  float    v8f;
typedef __attribute__((ext_vector_type(4)))  float    v4f;

// ---- WMMA fragment loaders (CDNA5 ISA 7.12.2 layouts, wave32) -------------

// A-matrix 16x32 f16 (MxK): lane m = lane&15; halfs 0..7 = K[off..off+7],
// halfs 8..15 = K[off+16..off+23], off = (lane<16)?0:8. Source row-major, K contiguous.
__device__ __forceinline__ v16h load_a_f16(const _Float16* __restrict__ A, size_t ld,
                                           int row_base, int k0, int lane) {
  const int m   = lane & 15;
  const int off = (lane < 16) ? 0 : 8;
  const _Float16* p = A + (size_t)(row_base + m) * ld + (size_t)(k0 + off);
  v8h lo = *(const v8h*)p;
  v8h hi = *(const v8h*)(p + 16);
  v16h r;
#pragma unroll
  for (int i = 0; i < 8; ++i) { r[i] = lo[i]; r[8 + i] = hi[i]; }
  return r;
}

// Same A fragment but source is f32 (converted to f16 on load).
__device__ __forceinline__ v16h load_a_f32(const float* __restrict__ A, size_t ld,
                                           int row_base, int k0, int lane) {
  const int m   = lane & 15;
  const int off = (lane < 16) ? 0 : 8;
  const float* p = A + (size_t)(row_base + m) * ld + (size_t)(k0 + off);
  v4f a0 = *(const v4f*)p;
  v4f a1 = *(const v4f*)(p + 4);
  v4f b0 = *(const v4f*)(p + 16);
  v4f b1 = *(const v4f*)(p + 20);
  v16h r;
#pragma unroll
  for (int i = 0; i < 4; ++i) {
    r[i]      = (_Float16)a0[i];
    r[4 + i]  = (_Float16)a1[i];
    r[8 + i]  = (_Float16)b0[i];
    r[12 + i] = (_Float16)b1[i];
  }
  return r;
}

// B-matrix 32x16 f16 (KxN): lane n = lane&15; halfs 0..7 = K[koff..koff+7],
// halfs 8..15 = K[koff+16..koff+23], koff = (lane<16)?0:8. Source "B^T" row-major [N][K], K contiguous.
__device__ __forceinline__ v16h load_b_f16(const _Float16* __restrict__ Bt, size_t ld,
                                           int col_base, int k0, int lane) {
  const int n    = lane & 15;
  const int koff = (lane < 16) ? 0 : 8;
  const _Float16* p = Bt + (size_t)(col_base + n) * ld + (size_t)(k0 + koff);
  v8h lo = *(const v8h*)p;
  v8h hi = *(const v8h*)(p + 16);
  v16h r;
#pragma unroll
  for (int i = 0; i < 8; ++i) { r[i] = lo[i]; r[8 + i] = hi[i]; }
  return r;
}

#define WMMA_F16(a, b, c) \
  __builtin_amdgcn_wmma_f32_16x16x32_f16(false, (a), false, (b), (short)0, (c), false, false)

// ---- converts -------------------------------------------------------------

__global__ __launch_bounds__(256) void k_cvt(const float* __restrict__ in,
                                             _Float16* __restrict__ out, size_t n) {
  size_t i = (size_t)blockIdx.x * 256 + threadIdx.x;
  if (i < n) out[i] = (_Float16)in[i];
}

// hidT[b][d][s] = hidden[b][s][d], f16
__global__ __launch_bounds__(256) void k_transpose_cvt(const float* __restrict__ hid,
                                                       _Float16* __restrict__ hidT) {
  size_t i = (size_t)blockIdx.x * 256 + threadIdx.x;  // over B*D*S
  const int s = (int)(i % S_);
  const int d = (int)((i / S_) % D_);
  const int b = (int)(i / ((size_t)S_ * D_));
  hidT[i] = (_Float16)hid[((size_t)b * S_ + s) * D_ + d];
}

// ---- GEMM 1: proj = tanh(hidden @ W^T + bias), f16 out --------------------
// M = B*S = 4096, N = E = 768, K = D = 768. 4 waves/block (in N), wave tile 32x64.
__global__ __launch_bounds__(128) void k_proj(const _Float16* __restrict__ hid_h,
                                              const _Float16* __restrict__ w_h,
                                              const float* __restrict__ bias,
                                              _Float16* __restrict__ proj_h) {
  const int lane = threadIdx.x & 31;
  const int wave = threadIdx.x >> 5;
  const int row_base = blockIdx.y * 32;                 // m (b*S+s)
  const int col_base = blockIdx.x * 256 + wave * 64;    // e
  v8f acc[2][4] = {};
  for (int k0 = 0; k0 < D_; k0 += 32) {
    if (k0 + 32 < D_)
      __builtin_prefetch(hid_h + (size_t)(row_base + (lane & 15)) * D_ + k0 + 32, 0, 1);
    v16h a0 = load_a_f16(hid_h, D_, row_base,      k0, lane);
    v16h a1 = load_a_f16(hid_h, D_, row_base + 16, k0, lane);
#pragma unroll
    for (int j = 0; j < 4; ++j) {
      v16h b = load_b_f16(w_h, D_, col_base + 16 * j, k0, lane);
      acc[0][j] = WMMA_F16(a0, b, acc[0][j]);
      acc[1][j] = WMMA_F16(a1, b, acc[1][j]);
    }
  }
  const int n = lane & 15;
  const int mo = (lane < 16) ? 0 : 8;
#pragma unroll
  for (int j = 0; j < 4; ++j) {
    const int e = col_base + 16 * j + n;
    const float bv = bias[e];
#pragma unroll
    for (int i = 0; i < 2; ++i)
#pragma unroll
      for (int r = 0; r < 8; ++r) {
        const int m = row_base + 16 * i + mo + r;
        proj_h[(size_t)m * E_ + e] = (_Float16)tanhf(acc[i][j][r] + bv);
      }
  }
}

// ---- GEMM 2: scores[b][l][s] = label_feat @ proj[b]^T, f32 out ------------
// per batch: M = L = 4000, N = S = 512, K = E = 768. 8 waves/block (in N),
// wave tile 32x64 -> one block spans all of S.
__global__ __launch_bounds__(256) void k_scores(const _Float16* __restrict__ lab_h,
                                                const _Float16* __restrict__ proj_h,
                                                float* __restrict__ scores) {
  const int bz = blockIdx.y;
  const int lane = threadIdx.x & 31;
  const int wave = threadIdx.x >> 5;
  const int row_base = blockIdx.x * 32;   // l
  const int col_base = wave * 64;         // s
  const _Float16* Bt = proj_h + (size_t)bz * S_ * E_;
  float* C = scores + (size_t)bz * L_ * S_;
  v8f acc[2][4] = {};
  for (int k0 = 0; k0 < E_; k0 += 32) {
    if (k0 + 32 < E_)
      __builtin_prefetch(lab_h + (size_t)(row_base + (lane & 15)) * E_ + k0 + 32, 0, 1);
    v16h a0 = load_a_f16(lab_h, E_, row_base,      k0, lane);
    v16h a1 = load_a_f16(lab_h, E_, row_base + 16, k0, lane);
#pragma unroll
    for (int j = 0; j < 4; ++j) {
      v16h b = load_b_f16(Bt, E_, col_base + 16 * j, k0, lane);
      acc[0][j] = WMMA_F16(a0, b, acc[0][j]);
      acc[1][j] = WMMA_F16(a1, b, acc[1][j]);
    }
  }
  const int n = lane & 15;
  const int mo = (lane < 16) ? 0 : 8;
#pragma unroll
  for (int i = 0; i < 2; ++i)
#pragma unroll
    for (int j = 0; j < 4; ++j)
#pragma unroll
      for (int r = 0; r < 8; ++r)
        C[(size_t)(row_base + 16 * i + mo + r) * S_ + col_base + 16 * j + n] = acc[i][j][r];
}

// ---- softmax over L (dim=1) ----------------------------------------------
// Pass 1: per (b,s) column, online max/sum. Block = 16 s-cols x 16 l-threads.
__global__ __launch_bounds__(256) void k_softmax_stats(const float* __restrict__ scores,
                                                       float* __restrict__ rowmax,
                                                       float* __restrict__ rowinv) {
  const int tx = threadIdx.x & 15;   // s lane
  const int ty = threadIdx.x >> 4;   // l group
  const int b  = blockIdx.y;
  const int s  = blockIdx.x * 16 + tx;
  const float* col = scores + (size_t)b * L_ * S_ + s;
  float m = -1e30f, sum = 0.f;
  for (int l = ty; l < L_; l += 16) {
    float x = col[(size_t)l * S_];
    if (x > m) { sum = sum * __expf(m - x) + 1.f; m = x; }
    else       { sum += __expf(x - m); }
  }
  __shared__ float sm[256], ss[256];
  sm[threadIdx.x] = m; ss[threadIdx.x] = sum;
  __syncthreads();
  for (int st = 8; st >= 1; st >>= 1) {
    if (ty < st) {
      float m1 = sm[ty * 16 + tx],        s1 = ss[ty * 16 + tx];
      float m2 = sm[(ty + st) * 16 + tx], s2 = ss[(ty + st) * 16 + tx];
      float mn = fmaxf(m1, m2);
      sm[ty * 16 + tx] = mn;
      ss[ty * 16 + tx] = s1 * __expf(m1 - mn) + s2 * __expf(m2 - mn);
    }
    __syncthreads();
  }
  if (ty == 0) {
    rowmax[b * S_ + s] = sm[tx];
    rowinv[b * S_ + s] = 1.f / ss[tx];
  }
}

// Pass 2: normalize in place.
__global__ __launch_bounds__(256) void k_softmax_write(float* __restrict__ weights,
                                                       const float* __restrict__ rowmax,
                                                       const float* __restrict__ rowinv) {
  size_t idx = (size_t)blockIdx.x * 256 + threadIdx.x;  // over B*L*S (exact)
  const int s = (int)(idx % S_);
  const int b = (int)(idx / ((size_t)L_ * S_));
  const int c = b * S_ + s;
  weights[idx] = __expf(weights[idx] - rowmax[c]) * rowinv[c];
}

// ---- GEMM 3: attn[b][l][d] = weights[b] @ hidden[b], f32 out --------------
// per batch: M = L = 4000, N = D = 768, K = S = 512. 4 waves/block (in N),
// wave tile 32x64. A (weights) is f32, converted to f16 on load.
__global__ __launch_bounds__(128) void k_attn(const float* __restrict__ weights,
                                              const _Float16* __restrict__ hidT,
                                              float* __restrict__ attn) {
  const int bz = blockIdx.z;
  const int lane = threadIdx.x & 31;
  const int wave = threadIdx.x >> 5;
  const int row_base = blockIdx.y * 32;                 // l
  const int col_base = blockIdx.x * 256 + wave * 64;    // d
  const float* A = weights + (size_t)bz * L_ * S_;
  const _Float16* Bt = hidT + (size_t)bz * D_ * S_;
  v8f acc[2][4] = {};
  for (int k0 = 0; k0 < S_; k0 += 32) {
    if (k0 + 32 < S_)
      __builtin_prefetch(A + (size_t)(row_base + (lane & 15)) * S_ + k0 + 32, 0, 1);
    v16h a0 = load_a_f32(A, S_, row_base,      k0, lane);
    v16h a1 = load_a_f32(A, S_, row_base + 16, k0, lane);
#pragma unroll
    for (int j = 0; j < 4; ++j) {
      v16h b = load_b_f16(Bt, S_, col_base + 16 * j, k0, lane);
      acc[0][j] = WMMA_F16(a0, b, acc[0][j]);
      acc[1][j] = WMMA_F16(a1, b, acc[1][j]);
    }
  }
  const int n = lane & 15;
  const int mo = (lane < 16) ? 0 : 8;
#pragma unroll
  for (int i = 0; i < 2; ++i)
#pragma unroll
    for (int j = 0; j < 4; ++j)
#pragma unroll
      for (int r = 0; r < 8; ++r)
        attn[((size_t)bz * L_ + row_base + 16 * i + mo + r) * D_ + col_base + 16 * j + n]
            = acc[i][j][r];
}

// ---- launch ---------------------------------------------------------------

extern "C" void kernel_launch(void* const* d_in, const int* in_sizes, int n_in,
                              void* d_out, int out_size, void* d_ws, size_t ws_size,
                              hipStream_t stream) {
  (void)in_sizes; (void)n_in; (void)out_size; (void)ws_size;
  const float* hid  = (const float*)d_in[0];  // [B,S,D]
  const float* lab  = (const float*)d_in[1];  // [L,E]
  const float* W    = (const float*)d_in[2];  // [E,D]
  const float* bias = (const float*)d_in[3];  // [E]

  float* weights = (float*)d_out;                         // [B,L,S]
  float* attn    = (float*)d_out + (size_t)B_ * L_ * S_;  // [B,L,D]

  char* ws = (char*)d_ws;
  size_t off = 0;
  auto alloc = [&](size_t bytes) -> void* {
    void* p = ws + off;
    off += (bytes + 255) & ~(size_t)255;
    return p;
  };
  _Float16* hid_h  = (_Float16*)alloc((size_t)B_ * S_ * D_ * 2);  // [B*S][D]
  _Float16* hidT_h = (_Float16*)alloc((size_t)B_ * D_ * S_ * 2);  // [B][D][S]
  _Float16* lab_h  = (_Float16*)alloc((size_t)L_ * E_ * 2);       // [L][E]
  _Float16* w_h    = (_Float16*)alloc((size_t)E_ * D_ * 2);       // [E][D]
  _Float16* proj_h = (_Float16*)alloc((size_t)B_ * S_ * E_ * 2);  // [B*S][E]
  float*    rowmax = (float*)alloc((size_t)B_ * S_ * 4);
  float*    rowinv = (float*)alloc((size_t)B_ * S_ * 4);

  { size_t n = (size_t)B_ * S_ * D_; k_cvt<<<(unsigned)((n + 255) / 256), 256, 0, stream>>>(hid, hid_h, n); }
  { size_t n = (size_t)L_ * E_;      k_cvt<<<(unsigned)((n + 255) / 256), 256, 0, stream>>>(lab, lab_h, n); }
  { size_t n = (size_t)E_ * D_;      k_cvt<<<(unsigned)((n + 255) / 256), 256, 0, stream>>>(W, w_h, n); }
  { size_t n = (size_t)B_ * D_ * S_; k_transpose_cvt<<<(unsigned)((n + 255) / 256), 256, 0, stream>>>(hid, hidT_h); }

  dim3 gp(E_ / 256, (B_ * S_) / 32, 1);
  k_proj<<<gp, 128, 0, stream>>>(hid_h, w_h, bias, proj_h);

  dim3 gs(L_ / 32, B_);
  k_scores<<<gs, 256, 0, stream>>>(lab_h, proj_h, weights);

  dim3 g1(S_ / 16, B_);
  k_softmax_stats<<<g1, 256, 0, stream>>>(weights, rowmax, rowinv);

  size_t tot = (size_t)B_ * L_ * S_;  // divisible by 256
  k_softmax_write<<<(unsigned)(tot / 256), 256, 0, stream>>>(weights, rowmax, rowinv);

  dim3 ga(D_ / 256, L_ / 32, B_);
  k_attn<<<ga, 128, 0, stream>>>(weights, hidT_h, attn);
}
